// Qwen3SeqParallelDecoderLayer_27951647162502
// MI455X (gfx1250) — matrix operand order repaired and checked by reading, not compiled, not run on hardware
//
#include <hip/hip_runtime.h>
#include <hip/hip_bf16.h>
#include <math.h>

// ---------------- model constants ----------------
#define HIDDEN  2048
#define NHEADS  16
#define NKV     8
#define HDIM    128
#define QDIM    2048        // NHEADS*HDIM
#define KVDIM   1024        // NKV*HDIM
#define QKVDIM  4096        // QDIM + 2*KVDIM
#define INTER   6144
#define SEQ     2048

typedef __attribute__((ext_vector_type(16))) __bf16 v16bf;
typedef __attribute__((ext_vector_type(8)))  float  v8f;
typedef __attribute__((ext_vector_type(4)))  unsigned int v4u;
typedef __attribute__((ext_vector_type(8)))  int v8i;
typedef __attribute__((ext_vector_type(4)))  int v4i;

#if defined(__has_builtin)
#  if __has_builtin(__builtin_amdgcn_tensor_load_to_lds)
#    define HAVE_TDM 1
#  endif
#endif
#ifndef HAVE_TDM
#  define HAVE_TDM 0
#endif

__device__ __forceinline__ unsigned short f2bf(float f) {
  unsigned int u = __float_as_uint(f);
  u += 0x7FFFu + ((u >> 16) & 1u);       // round-to-nearest-even
  return (unsigned short)(u >> 16);
}

union FragB { v16bf bf; unsigned short us[16]; uint4 q[2]; };

static __device__ __forceinline__ v8f wmma_bf16(const FragB& a, const FragB& b, v8f c) {
  // D = A(16x32 bf16) * B(32x16 bf16) + C(16x16 f32)
  return __builtin_amdgcn_wmma_f32_16x16x32_bf16(false, a.bf, false, b.bf,
                                                 (short)0, c, false, false);
}

// ---- CDNA5 async global->LDS copy (ASYNCcnt tracked), CDNA5 ISA ----
__device__ __forceinline__ void async_ld_b128(unsigned lds_off, const void* g) {
  asm volatile("global_load_async_to_lds_b128 %0, %1, off"
               :: "v"(lds_off), "v"(g) : "memory");
}
__device__ __forceinline__ void wait_async0() {
  asm volatile("s_wait_asynccnt 0" ::: "memory");
}
__device__ __forceinline__ void wait_async4() {
  asm volatile("s_wait_asynccnt 4" ::: "memory");
}

#if HAVE_TDM
// ---- CDNA5 Tensor Data Mover: 2D tile load Global->LDS (CDNA5 ISA-8)
// Tile (tile_d1 rows) x (tile_d0 elems of 2B), row stride K elems in memory.
// LDS pad: every 16 DWORDs (64B = 32 halves) insert 4 DWORDs (16B = 8 halves)
// -> rows land at stride 40 halves, matching the compute-side layout.
__device__ __forceinline__ void tdm_load_2d(
    unsigned lds_addr, const void* gaddr,
    unsigned tensor_d0, unsigned tensor_d1,
    unsigned tile_d0, unsigned tile_d1, unsigned stride_elems)
{
  unsigned long long ga = (unsigned long long)(uintptr_t)gaddr;
  v4u g0;
  g0.x = 1u;                                        // count=1, user mode
  g0.y = lds_addr;                                  // lds_addr [63:32]
  g0.z = (unsigned)(ga & 0xFFFFFFFFu);              // global_addr [95:64]
  g0.w = (unsigned)((ga >> 32) & 0x01FFFFFFu)       // global_addr [120:96]
       | (2u << 30);                                // type=2 ("image")
  v8i g1;
  g1[0] = (int)((1u << 16)                          // data_size = 2 bytes
              | (1u << 20)                          // pad_enable
              | (3u << 22)                          // pad_interval: 16 DWORDs
              | (3u << 25));                        // pad_amount: 4 DWORDs
  g1[1] = (int)((tensor_d0 & 0xFFFFu) << 16);               // tensor_dim0 lo
  g1[2] = (int)(((tensor_d0 >> 16) & 0xFFFFu)
              | ((tensor_d1 & 0xFFFFu) << 16));             // dim0 hi | dim1 lo
  g1[3] = (int)(((tensor_d1 >> 16) & 0xFFFFu)
              | ((tile_d0 & 0xFFFFu) << 16));               // dim1 hi | tile_dim0
  g1[4] = (int)(tile_d1 & 0xFFFFu);                         // tile_dim1, tile_dim2=0
  g1[5] = (int)stride_elems;                                // tensor_dim0_stride lo
  g1[6] = 0;                                                // stride hi, dim1_stride lo
  g1[7] = 0;
  v4i zz = {0, 0, 0, 0};
#if defined(__clang_major__) && __clang_major__ >= 23
  v8i zz8 = {0, 0, 0, 0, 0, 0, 0, 0};
  __builtin_amdgcn_tensor_load_to_lds(g0, g1, zz, zz, zz8, 0);
#else
  __builtin_amdgcn_tensor_load_to_lds(g0, g1, zz, zz, 0);
#endif
}
#endif

// =====================================================================
// fp32 -> bf16 elementwise convert (weights; n multiple of 4)
// =====================================================================
__global__ void __launch_bounds__(256) cvt_bf16_k(
    const float* __restrict__ in, unsigned short* __restrict__ out, int n4)
{
  const int t = blockIdx.x * 256 + threadIdx.x;
  if (t >= n4) return;
  float4 f = reinterpret_cast<const float4*>(in)[t];
  union { unsigned short us[4]; uint2 u2; } p;
  p.us[0] = f2bf(f.x); p.us[1] = f2bf(f.y);
  p.us[2] = f2bf(f.z); p.us[3] = f2bf(f.w);
  reinterpret_cast<uint2*>(out)[t] = p.u2;
}

// =====================================================================
// RMSNorm (fp32 in) -> bf16 out
// =====================================================================
__global__ void __launch_bounds__(256) rmsnorm_bf16_k(
    const float* __restrict__ x, const float* __restrict__ w,
    unsigned short* __restrict__ out, int H)
{
  __shared__ float red[8];
  const int row = blockIdx.x, tid = threadIdx.x;
  const float* xr = x + (size_t)row * H;
  float ss = 0.f;
  for (int i = tid; i < H; i += 256) { float v = xr[i]; ss += v * v; }
  for (int m = 16; m >= 1; m >>= 1) ss += __shfl_xor(ss, m, 32);
  if ((tid & 31) == 0) red[tid >> 5] = ss;
  __syncthreads();
  float tot = 0.f;
  #pragma unroll
  for (int i = 0; i < 8; i++) tot += red[i];
  const float rstd = rsqrtf(tot / (float)H + 1e-6f);
  unsigned short* orow = out + (size_t)row * H;
  for (int i = tid; i < H; i += 256) orow[i] = f2bf(xr[i] * rstd * w[i]);
}

// =====================================================================
// WMMA GEMM: C[M,N] = A[M,K](bf16) * B[N,K](bf16)^T (+resid), fp32 out.
// 128x128 block tile, 8 waves (4x2), K-step 32, double-buffered LDS
// filled by the Tensor Data Mover (one descriptor per tile), with a
// per-thread async-to-LDS fallback when the TDM builtin is unavailable.
// =====================================================================
#define BK     32
#define PAD_LD 40   // padded LDS row stride (halves): 80B, breaks 4-way bank conflicts

template<bool RESID>
__global__ void __launch_bounds__(256) gemm_bf16(
    const unsigned short* __restrict__ A, const unsigned short* __restrict__ B,
    float* __restrict__ C, const float* __restrict__ resid,
    int M, int N, int K)
{
  __shared__ __align__(16) unsigned short As[2][128 * PAD_LD];
  __shared__ __align__(16) unsigned short Bs[2][128 * PAD_LD];
  const int tid  = threadIdx.x;
  const int lane = tid & 31, wave = tid >> 5;
  const int m0 = blockIdx.y * 128, n0 = blockIdx.x * 128;
  const int wm = wave & 3, wn = wave >> 2;       // 4 waves in M, 2 in N
  const int nt = K / BK;

  const v8f z = {0.f,0.f,0.f,0.f,0.f,0.f,0.f,0.f};
  v8f acc[2][4];
  #pragma unroll
  for (int i = 0; i < 2; i++)
    #pragma unroll
    for (int j = 0; j < 4; j++) acc[i][j] = z;

#if HAVE_TDM
  const unsigned aBase[2] = { (unsigned)(uintptr_t)&As[0][0],
                              (unsigned)(uintptr_t)&As[1][0] };
  const unsigned bBase[2] = { (unsigned)(uintptr_t)&Bs[0][0],
                              (unsigned)(uintptr_t)&Bs[1][0] };
  auto issue = [&](int kt, int buf) {   // wave 0 only: one TDM op per tile
    tdm_load_2d(aBase[buf], A + (size_t)m0 * K + kt * BK, K, M - m0, BK, 128, K);
    tdm_load_2d(bBase[buf], B + (size_t)n0 * K + kt * BK, K, N - n0, BK, 128, K);
  };
  if (wave == 0) issue(0, 0);
#else
  const int lr = tid >> 1, ls = (tid & 1) * 16;  // tile-load row / 16-half segment
  const unsigned short* aptr = A + (size_t)(m0 + lr) * K + ls;
  const unsigned short* bptr = B + (size_t)(n0 + lr) * K + ls;
  const unsigned aL[2] = { (unsigned)(uintptr_t)&As[0][lr * PAD_LD + ls],
                           (unsigned)(uintptr_t)&As[1][lr * PAD_LD + ls] };
  const unsigned bL[2] = { (unsigned)(uintptr_t)&Bs[0][lr * PAD_LD + ls],
                           (unsigned)(uintptr_t)&Bs[1][lr * PAD_LD + ls] };
  auto issue = [&](int kt, int buf) {
    const int k0 = kt * BK;
    async_ld_b128(aL[buf],      aptr + k0);
    async_ld_b128(aL[buf] + 16, aptr + k0 + 8);
    async_ld_b128(bL[buf],      bptr + k0);
    async_ld_b128(bL[buf] + 16, bptr + k0 + 8);
  };
  issue(0, 0);
#endif

  for (int kt = 0; kt < nt; ++kt) {
    const int cur = kt & 1;
    const bool more = (kt + 1 < nt);
#if HAVE_TDM
    if (wave == 0) {
      if (more) { issue(kt + 1, cur ^ 1);                    // prefetch next tile
                  __builtin_amdgcn_s_wait_tensorcnt((short)2); } // tile kt landed
      else      { __builtin_amdgcn_s_wait_tensorcnt((short)0); }
    }
#else
    if (more) { issue(kt + 1, cur ^ 1); wait_async4(); }
    else      { wait_async0(); }
#endif
    __syncthreads();

    FragB af[2], bf2[4];
    #pragma unroll
    for (int mf = 0; mf < 2; ++mf) {   // A frag: lane=row, chunks at kb and kb+16
      const unsigned short* p =
          &As[cur][(wm*32 + mf*16 + (lane & 15)) * PAD_LD + ((lane >> 4) << 3)];
      af[mf].q[0] = *reinterpret_cast<const uint4*>(p);
      af[mf].q[1] = *reinterpret_cast<const uint4*>(p + 16);
    }
    #pragma unroll
    for (int nf = 0; nf < 4; ++nf) {   // B frag: lane=col, chunks at kb and kb+16
      const unsigned short* p =
          &Bs[cur][(wn*64 + nf*16 + (lane & 15)) * PAD_LD + ((lane >> 4) << 3)];
      bf2[nf].q[0] = *reinterpret_cast<const uint4*>(p);
      bf2[nf].q[1] = *reinterpret_cast<const uint4*>(p + 16);
    }
    #pragma unroll
    for (int mf = 0; mf < 2; ++mf)
      #pragma unroll
      for (int nf = 0; nf < 4; ++nf)
        acc[mf][nf] = wmma_bf16(af[mf], bf2[nf], acc[mf][nf]);
    __syncthreads();   // all waves done with buf[cur] before it is refilled
  }

  const int cm = (lane >> 4) << 3;   // C layout: M = r + cm, N = lane&15
  const int cn = lane & 15;
  #pragma unroll
  for (int mf = 0; mf < 2; ++mf)
    #pragma unroll
    for (int nf = 0; nf < 4; ++nf) {
      const int row = m0 + wm*32 + mf*16 + cm;
      const int col = n0 + wn*64 + nf*16 + cn;
      size_t idx = (size_t)row * N + col;
      #pragma unroll
      for (int r = 0; r < 8; r++, idx += N) {
        float v = acc[mf][nf][r];
        if constexpr (RESID) v += resid[idx];
        C[idx] = v;
      }
    }
}

// =====================================================================
// QK rmsnorm + RoPE (+ 1/sqrt(d) folded into Q) + V bf16 convert
// grid (S, 32): y<16 -> Q head, 16..23 -> K head, 24..31 -> V head
// =====================================================================
__global__ void __launch_bounds__(128) qk_rope_k(
    const float* __restrict__ qkv, const int* __restrict__ pos,
    const float* __restrict__ qw, const float* __restrict__ kw,
    unsigned short* __restrict__ qo, unsigned short* __restrict__ ko,
    unsigned short* __restrict__ vo)
{
  __shared__ float red[4];
  __shared__ float tv[HDIM];
  const int s = blockIdx.x, y = blockIdx.y, d = threadIdx.x;

  if (y >= 24) {  // V: straight bf16 convert
    const int h = y - 24;
    float v = qkv[(size_t)s*QKVDIM + QDIM + KVDIM + (size_t)h*HDIM + d];
    vo[((size_t)s*NKV + h)*HDIM + d] = f2bf(v);
    return;
  }
  const bool isq = (y < NHEADS);
  const int  h   = isq ? y : (y - NHEADS);
  const size_t base = (size_t)s*QKVDIM + (isq ? 0 : QDIM) + (size_t)h*HDIM;
  const float* nw = isq ? qw : kw;

  const float xv = qkv[base + d];
  float ss = xv * xv;
  for (int m = 16; m >= 1; m >>= 1) ss += __shfl_xor(ss, m, 32);
  if ((d & 31) == 0) red[d >> 5] = ss;
  __syncthreads();
  const float tot  = red[0] + red[1] + red[2] + red[3];
  const float rstd = rsqrtf(tot / (float)HDIM + 1e-6f);
  tv[d] = xv * nw[d];
  __syncthreads();
  const float val   = tv[d]      * rstd;
  const float other = tv[d ^ 64] * rstd;

  const int p = pos[s];
  const int j = d & 63;
  const float freq = (float)p * __powf(10000.f, -(float)(2*j) * (1.f/128.f));
  float sn, c;
  __sincosf(freq, &sn, &c);
  float r = (d < 64) ? (val*c - other*sn) : (val*c + other*sn);
  if (isq) {
    r *= 0.08838834764831845f;  // 1/sqrt(128), folded so scores need no scale
    qo[((size_t)s*NHEADS + h)*HDIM + d] = f2bf(r);
  } else {
    ko[((size_t)s*NKV + h)*HDIM + d] = f2bf(r);
  }
}

// =====================================================================
// WMMA flash attention (causal, GQA rep=2).
// Block = 4 waves = 64 query rows for one head; key tiles of 32.
// K tile staged with async-to-LDS; V tile transposed through registers.
// =====================================================================
#define KLD 136   // K tile LDS stride (halves): 128 + 8 pad
#define VLD 40    // V^T / P LDS stride (halves): 32 + 8 pad

__global__ void __launch_bounds__(128) flash_attn_k(
    const unsigned short* __restrict__ Q,   // [S,16,128] bf16, pre-scaled
    const unsigned short* __restrict__ Kb,  // [S,8,128]  bf16
    const unsigned short* __restrict__ Vb,  // [S,8,128]  bf16
    unsigned short* __restrict__ O,         // [S,2048]   bf16
    int S)
{
  __shared__ __align__(16) unsigned short Ks[32 * KLD];    // keys x d
  __shared__ __align__(16) unsigned short Vt[HDIM * VLD];  // d x keys
  __shared__ __align__(16) unsigned short Ps[4][16 * VLD]; // per-wave P tile 16x32

  const int tid = threadIdx.x, lane = tid & 31, wave = tid >> 5;
  const int h = blockIdx.y, kvh = h >> 1;
  const int qb = blockIdx.x * 64 + wave * 16;

  // Q fragments: 4 K-chunks of 32 over head_dim
  FragB qf[4];
  {
    const unsigned short* qp = Q + ((size_t)(qb + (lane & 15)) * NHEADS + h) * HDIM;
    #pragma unroll
    for (int kk = 0; kk < 4; ++kk) {
      const int off = kk*32 + ((lane >> 4) << 3);
      qf[kk].q[0] = *reinterpret_cast<const uint4*>(qp + off);
      qf[kk].q[1] = *reinterpret_cast<const uint4*>(qp + off + 16);
    }
  }

  const v8f z = {0.f,0.f,0.f,0.f,0.f,0.f,0.f,0.f};
  v8f oacc[8];
  #pragma unroll
  for (int i = 0; i < 8; i++) oacc[i] = z;
  float mrow[8], lrow[8];
  #pragma unroll
  for (int r = 0; r < 8; r++) { mrow[r] = -1e30f; lrow[r] = 0.f; }

  const int lkey = tid >> 2, lseg = (tid & 3) * 32;   // cooperative-load coords
  const unsigned ksl = (unsigned)(uintptr_t)&Ks[lkey * KLD + lseg];
  const int kend = blockIdx.x * 64 + 64;              // block needs keys < kend

  for (int kt = 0; kt < kend; kt += 32) {
    {  // K tile: 64B per thread via 4 async b128 copies
      const unsigned short* kp = Kb + ((size_t)(kt + lkey)*NKV + kvh)*HDIM + lseg;
      async_ld_b128(ksl,      kp);
      async_ld_b128(ksl + 16, kp + 8);
      async_ld_b128(ksl + 32, kp + 16);
      async_ld_b128(ksl + 48, kp + 24);
      // V tile: vector global read, transposed scalar LDS writes
      const uint4* vp = reinterpret_cast<const uint4*>(
          Vb + ((size_t)(kt + lkey)*NKV + kvh)*HDIM + lseg);
      union { uint4 q[4]; unsigned short us[32]; } vv;
      vv.q[0] = vp[0]; vv.q[1] = vp[1]; vv.q[2] = vp[2]; vv.q[3] = vp[3];
      #pragma unroll
      for (int i = 0; i < 32; i++) Vt[(lseg + i)*VLD + lkey] = vv.us[i];
    }
    wait_async0();
    __syncthreads();

    if (kt < qb + 16) {   // this wave has un-masked work in the tile
      // ---- scores: S = Q * K^T (contraction over d=128) ----
      v8f sf[2]; sf[0] = z; sf[1] = z;
      #pragma unroll
      for (int kf = 0; kf < 2; ++kf)
        #pragma unroll
        for (int kk = 0; kk < 4; ++kk) {
          FragB kb_;
          const unsigned short* p =
              &Ks[(kf*16 + (lane & 15))*KLD + kk*32 + ((lane >> 4) << 3)];
          kb_.q[0] = *reinterpret_cast<const uint4*>(p);
          kb_.q[1] = *reinterpret_cast<const uint4*>(p + 16);
          sf[kf] = wmma_bf16(qf[kk], kb_, sf[kf]);
        }

      // ---- causal mask + online softmax (rows live in r + lane-half) ----
      const int key0 = kt + (lane & 15);
      const int key1 = key0 + 16;
      const int qr0  = qb + ((lane >> 4) << 3);
      float p0[8], p1[8];
      #pragma unroll
      for (int r = 0; r < 8; r++) {
        const int qr = qr0 + r;
        float s0 = (key0 <= qr) ? sf[0][r] : -1e30f;
        float s1 = (key1 <= qr) ? sf[1][r] : -1e30f;
        float rm = fmaxf(s0, s1);
        rm = fmaxf(rm, __shfl_xor(rm, 1, 32));
        rm = fmaxf(rm, __shfl_xor(rm, 2, 32));
        rm = fmaxf(rm, __shfl_xor(rm, 4, 32));
        rm = fmaxf(rm, __shfl_xor(rm, 8, 32));
        const float mn   = fmaxf(mrow[r], rm);
        const float corr = __expf(mrow[r] - mn);
        const float e0 = __expf(s0 - mn);
        const float e1 = __expf(s1 - mn);
        float rs = e0 + e1;
        rs += __shfl_xor(rs, 1, 32);
        rs += __shfl_xor(rs, 2, 32);
        rs += __shfl_xor(rs, 4, 32);
        rs += __shfl_xor(rs, 8, 32);
        lrow[r] = lrow[r]*corr + rs;
        mrow[r] = mn;
        p0[r] = e0; p1[r] = e1;
        #pragma unroll
        for (int f = 0; f < 8; f++) oacc[f][r] *= corr;
      }

      // ---- P: C-layout -> A-layout via per-wave LDS ----
      unsigned short* pw = Ps[wave];
      #pragma unroll
      for (int r = 0; r < 8; r++) {
        const int prow = ((lane >> 4) << 3) + r;
        pw[prow*VLD + (lane & 15)]      = f2bf(p0[r]);
        pw[prow*VLD + 16 + (lane & 15)] = f2bf(p1[r]);
      }
      FragB pf;
      {
        const unsigned short* pp = &pw[(lane & 15)*VLD + ((lane >> 4) << 3)];
        pf.q[0] = *reinterpret_cast<const uint4*>(pp);
        pf.q[1] = *reinterpret_cast<const uint4*>(pp + 16);
      }

      // ---- O += P * V (contraction over 32 keys) ----
      #pragma unroll
      for (int nf = 0; nf < 8; ++nf) {
        FragB vf;
        const unsigned short* p =
            &Vt[(nf*16 + (lane & 15))*VLD + ((lane >> 4) << 3)];
        vf.q[0] = *reinterpret_cast<const uint4*>(p);
        vf.q[1] = *reinterpret_cast<const uint4*>(p + 16);
        oacc[nf] = wmma_bf16(pf, vf, oacc[nf]);
      }
    }
    __syncthreads();
  }

  // ---- finalize: divide by row sums, store bf16 [s, h*128 + d] ----
  const int cn = lane & 15, cm = (lane >> 4) << 3;
  #pragma unroll
  for (int r = 0; r < 8; r++) lrow[r] = 1.f / lrow[r];
  #pragma unroll
  for (int nf = 0; nf < 8; ++nf) {
    const int col = h*HDIM + nf*16 + cn;
    #pragma unroll
    for (int r = 0; r < 8; r++) {
      const int row = qb + cm + r;
      O[(size_t)row*QDIM + col] = f2bf(oacc[nf][r] * lrow[r]);
    }
  }
}

// =====================================================================
// SwiGLU: silu(gate) * up  -> bf16
// =====================================================================
__global__ void __launch_bounds__(256) swiglu_k(
    const float* __restrict__ gu, unsigned short* __restrict__ out, int total)
{
  const int t = blockIdx.x * 256 + threadIdx.x;
  if (t >= total) return;
  const int s = t / INTER, i = t % INTER;
  const float g = gu[(size_t)s*(2*INTER) + i];
  const float u = gu[(size_t)s*(2*INTER) + INTER + i];
  const float si = g / (1.f + __expf(-g));
  out[t] = f2bf(si * u);
}

// =====================================================================
// Launch
// =====================================================================
extern "C" void kernel_launch(void* const* d_in, const int* in_sizes, int n_in,
                              void* d_out, int out_size, void* d_ws, size_t ws_size,
                              hipStream_t stream)
{
  const float* x         = (const float*)d_in[0];
  const int*   pos       = (const int*)  d_in[1];
  const float* in_ln_w   = (const float*)d_in[2];
  const float* qkv_w     = (const float*)d_in[3];
  const float* q_norm_w  = (const float*)d_in[4];
  const float* k_norm_w  = (const float*)d_in[5];
  const float* o_w       = (const float*)d_in[6];
  const float* post_ln_w = (const float*)d_in[7];
  const float* gu_w      = (const float*)d_in[8];
  const float* down_w    = (const float*)d_in[9];

  float* out_h   = (float*)d_out;                           // output[0]
  float* out_res = (float*)d_out + (size_t)SEQ * HIDDEN;    // output[1]

  char* ws = (char*)d_ws;
  const size_t MB = 1024ull * 1024ull;
  unsigned short* wbf     = (unsigned short*)(ws + 0);        // 48 MB  reused per GEMM
  unsigned short* h_bf    = (unsigned short*)(ws + 48*MB);    // 8 MB   live 1-2
  float*          qkv     = (float*)         (ws + 56*MB);    // 32 MB  live 2-3
  unsigned short* q_bf    = (unsigned short*)(ws + 88*MB);    // 8 MB   live 3-4
  unsigned short* k_bf    = (unsigned short*)(ws + 96*MB);    // 4 MB   live 3-4
  unsigned short* v_bf    = (unsigned short*)(ws + 100*MB);   // 4 MB   live 3-4
  unsigned short* attn_bf = (unsigned short*)(ws + 104*MB);   // 8 MB   live 4-5
  unsigned short* h2_bf   = (unsigned short*)(ws + 48*MB);    // reuse  live 6-7
  float*          gu      = (float*)         (ws + 56*MB);    // 96 MB  live 7-8 (reuse)
  unsigned short* act_bf  = (unsigned short*)(ws + 152*MB);   // 24 MB  live 8-9

  // 1. input rmsnorm -> bf16
  rmsnorm_bf16_k<<<SEQ, 256, 0, stream>>>(x, in_ln_w, h_bf, HIDDEN);
  // 2. QKV projection (bf16 WMMA, TDM-staged tiles)
  cvt_bf16_k<<<(QKVDIM*HIDDEN/4 + 255)/256, 256, 0, stream>>>(qkv_w, wbf, QKVDIM*HIDDEN/4);
  gemm_bf16<false><<<dim3(QKVDIM/128, SEQ/128), 256, 0, stream>>>(
      h_bf, wbf, qkv, nullptr, SEQ, QKVDIM, HIDDEN);
  // 3. QK norm + RoPE + scale, V convert
  qk_rope_k<<<dim3(SEQ, 32), 128, 0, stream>>>(
      qkv, pos, q_norm_w, k_norm_w, q_bf, k_bf, v_bf);
  // 4. causal flash attention (bf16 WMMA, async-to-LDS K tiles)
  flash_attn_k<<<dim3(SEQ/64, NHEADS), 128, 0, stream>>>(
      q_bf, k_bf, v_bf, attn_bf, SEQ);
  // 5. O projection + residual add -> residual output (fp32)
  cvt_bf16_k<<<(HIDDEN*QDIM/4 + 255)/256, 256, 0, stream>>>(o_w, wbf, HIDDEN*QDIM/4);
  gemm_bf16<true><<<dim3(HIDDEN/128, SEQ/128), 256, 0, stream>>>(
      attn_bf, wbf, out_res, x, SEQ, HIDDEN, QDIM);
  // 6. post-attention rmsnorm -> bf16
  rmsnorm_bf16_k<<<SEQ, 256, 0, stream>>>(out_res, post_ln_w, h2_bf, HIDDEN);
  // 7. gate_up projection
  cvt_bf16_k<<<(2*INTER*HIDDEN/4 + 255)/256, 256, 0, stream>>>(gu_w, wbf, 2*INTER*HIDDEN/4);
  gemm_bf16<false><<<dim3((2*INTER)/128, SEQ/128), 256, 0, stream>>>(
      h2_bf, wbf, gu, nullptr, SEQ, 2*INTER, HIDDEN);
  // 8. SwiGLU -> bf16
  swiglu_k<<<(SEQ*INTER + 255)/256, 256, 0, stream>>>(gu, act_bf, SEQ*INTER);
  // 9. down projection -> h output (fp32)
  cvt_bf16_k<<<(HIDDEN*INTER/4 + 255)/256, 256, 0, stream>>>(down_w, wbf, HIDDEN*INTER/4);
  gemm_bf16<false><<<dim3(HIDDEN/128, SEQ/128), 256, 0, stream>>>(
      act_bf, wbf, out_h, nullptr, SEQ, HIDDEN, INTER);
}
